// Stronger_CenterNet_54262616817682
// MI455X (gfx1250) — matrix operand order repaired and checked by reading, not compiled, not run on hardware
//
#include <hip/hip_runtime.h>
#include <hip/hip_bf16.h>
#include <math.h>

typedef __attribute__((ext_vector_type(16))) _Float16 v16h;
typedef __attribute__((ext_vector_type(8)))  float    v8f;

#define CDIV(a,b) (((a)+(b)-1)/(b))

static __device__ __forceinline__ int imin(int a, int b) { return a < b ? a : b; }

#define WMMA_F16(a, b, c) \
  __builtin_amdgcn_wmma_f32_16x16x32_f16(false, (a), false, (b), (short)0, (c), false, false)

// ---------------------------------------------------------------------------
// Fragment address helpers (CDNA5 WMMA 16x16x32 f16 layouts, wave32)
// A packed layout: [mt][kt][lane 0..31][16 halves]   (each frag = 512 halves)
// B packed layout: [nt][kt][lane 0..31][16 halves]
// B element (k,n): lane = (n&15) + 16*((k>>4)&1), half = k&15
// ---------------------------------------------------------------------------
static __device__ __forceinline__ size_t b_addr(int k, int n, int Kt) {
  int nt = n >> 4, kt = k >> 5;
  int lane = (n & 15) + (((k >> 4) & 1) << 4);
  int h = k & 15;
  return ((size_t)nt * Kt + kt) * 512 + (size_t)lane * 16 + h;
}

// ---------------------------------------------------------------------------
// Epilogue coefficients: out = acc * s[m] + t[m]  (optionally ReLU)
// ---------------------------------------------------------------------------
__global__ void make_coef_kernel(const float* bias, const float* gamma,
                                 const float* beta, const float* mean,
                                 const float* var, float* s, float* t,
                                 int M, int bn, float postmul) {
  int m = blockIdx.x * 256 + threadIdx.x;
  if (m >= M) return;
  if (bn) {
    float sc = gamma[m] * rsqrtf(var[m] + 1e-5f);
    s[m] = sc;
    t[m] = (bias[m] - mean[m]) * sc + beta[m];
  } else {
    s[m] = postmul;
    t[m] = postmul * bias[m];
  }
}

// ---------------------------------------------------------------------------
// Pack a row-major (M,K) fp32 weight matrix into f16 A-fragments.
// ---------------------------------------------------------------------------
__global__ void pack_a_kernel(const float* __restrict__ w,
                              _Float16* __restrict__ Ap,
                              int M, int K, int Mt, int Kt) {
  size_t idx = (size_t)blockIdx.x * 256 + threadIdx.x;
  size_t total = (size_t)Mt * Kt * 512;
  if (idx >= total) return;
  int h = (int)(idx & 15);
  int lane = (int)((idx >> 4) & 31);
  size_t frag = idx >> 9;
  int kt = (int)(frag % Kt);
  int mt = (int)(frag / Kt);
  int m = mt * 16 + (lane & 15);
  int g = lane >> 4;
  int k = kt * 32 + 16 * g + h;
  float v = (m < M && k < K) ? w[(size_t)m * K + k] : 0.0f;
  Ap[idx] = (_Float16)v;
}

// ---------------------------------------------------------------------------
// Standard im2col (KS=3,pad=1 or KS=1,pad=0) -> B fragments, f16.
// Square power-of-two images: H = W = 1<<lw.
// ---------------------------------------------------------------------------
template <int KS>
__global__ void pack_b_im2col_kernel(const float* __restrict__ x,
                                     _Float16* __restrict__ Bp,
                                     int C, int lw, int Kt) {
  const int W = 1 << lw;
  const int N = 1 << (2 * lw);
  const int K = C * KS * KS;
  const int pad = (KS == 3) ? 1 : 0;
  size_t idx = (size_t)blockIdx.x * 256 + threadIdx.x;
  if (idx >= (size_t)K * N) return;
  int n = (int)(idx & (N - 1));
  int k = (int)(idx >> (2 * lw));
  int c = k / (KS * KS);
  int tap = k % (KS * KS);
  int ty = tap / KS, tx = tap % KS;
  int y = n >> lw, xx = n & (W - 1);
  int iy = y + ty - pad, ix = xx + tx - pad;
  float v = 0.0f;
  if (iy >= 0 && iy < W && ix >= 0 && ix < W)
    v = x[(size_t)c * N + (iy << lw) + ix];
  Bp[b_addr(k, n, Kt)] = (_Float16)v;
}

// ---------------------------------------------------------------------------
// Deformable bilinear sampling * sigmoid(mask) -> B fragments, f16.
// K index = c*9 + tap (matches w.reshape(O, C*9)). H = W = 1<<lw.
// ---------------------------------------------------------------------------
__global__ void deform_pack_kernel(const float* __restrict__ x,
                                   const float* __restrict__ om,
                                   _Float16* __restrict__ Bp,
                                   int C, int lw, int Kt) {
  const int W = 1 << lw;
  const int N = 1 << (2 * lw);
  size_t idx = (size_t)blockIdx.x * 256 + threadIdx.x;
  if (idx >= (size_t)C * 9 * N) return;
  int n = (int)(idx & (N - 1));
  int k = (int)(idx >> (2 * lw));
  int c = k / 9, tap = k % 9;
  int y = n >> lw, xx = n & (W - 1);
  float dy = om[(size_t)(2 * tap) * N + n];
  float dx = om[(size_t)(2 * tap + 1) * N + n];
  float mk = om[(size_t)(18 + tap) * N + n];
  float mask = 1.0f / (1.0f + expf(-mk));
  float py = (float)y - 1.0f + (float)(tap / 3) + dy;
  float px = (float)xx - 1.0f + (float)(tap % 3) + dx;
  float fy0 = floorf(py), fx0 = floorf(px);
  int y0 = (int)fy0, x0 = (int)fx0;
  float fy = py - fy0, fx = px - fx0;
  const float* xc = x + (size_t)c * N;
  float s = 0.0f;
#pragma unroll
  for (int cy = 0; cy < 2; ++cy) {
#pragma unroll
    for (int cx = 0; cx < 2; ++cx) {
      int yi = y0 + cy, xi = x0 + cx;
      float wy = cy ? fy : (1.0f - fy);
      float wx = cx ? fx : (1.0f - fx);
      if (yi >= 0 && yi < W && xi >= 0 && xi < W)
        s += xc[(yi << lw) + xi] * (wy * wx);
    }
  }
  s *= mask;
  Bp[b_addr(c * 9 + tap, n, Kt)] = (_Float16)s;
}

// ---------------------------------------------------------------------------
// WMMA GEMM: out(MxN row-major) = A(MxK)*B(KxN), fused affine + ReLU.
// Block = 256 threads = 8 waves; tile 128 rows x 64 cols.
// B strip staged to LDS once per block (broadcast to 8 waves), manually
// unrolled x2 with STATIC double buffers + scalar (readfirstlane) wave
// predication so the WMMA path is free of exec-mask churn and reg rotation.
// Requires N % 64 == 0, K % 64 == 0-or-32 with Kt even (true here: Kt in
// {144,72,36,4,2}).
// ---------------------------------------------------------------------------
__global__ __launch_bounds__(256) void wmma_gemm_kernel(
    const _Float16* __restrict__ A, const _Float16* __restrict__ B,
    float* __restrict__ out, const float* __restrict__ cs,
    const float* __restrict__ ct, int M, int N, int Kt, int Mtiles,
    int relu) {
  __shared__ __attribute__((aligned(16))) _Float16 lds[2 * 2048];

  int tid = (int)threadIdx.x;
  int lane = tid & 31;
  int ntBase = blockIdx.x * 4;  // 4 n-tiles (64 columns) per block
  // wave-uniform m-tile id, forced into an SGPR -> scalar branches
  int mt = __builtin_amdgcn_readfirstlane(blockIdx.y * 8 + (tid >> 5));
  bool active = (mt < Mtiles);
  int mts = active ? mt : 0;

  // Cooperative B staging: 256 threads x 16B = 4KB = 4 frags of 512 halves.
  int st = tid >> 6;
  int so = (tid & 63) << 3;
  const _Float16* bsrc = B + ((size_t)(ntBase + st) * Kt) * 512 + so;
  _Float16* ld0 = lds + st * 512 + so;
  _Float16* ld1 = ld0 + 2048;

  const v16h* Af = (const v16h*)A + (size_t)mts * Kt * 32 + lane;
  const v16h* bl0 = (const v16h*)lds;
  const v16h* bl1 = (const v16h*)(lds + 2048);

  // Prologue: stage kt=0 -> buf0; prefetch B(kt=1) into regs; A(kt=0).
  *(uint4*)ld0 = *(const uint4*)bsrc;
  uint4 g = *(const uint4*)(bsrc + 512);  // Kt >= 2 always
  v16h a0 = {}, a1 = {};
  if (active) a0 = Af[0];
  __syncthreads();

  v8f acc[4] = {};
  for (int kt = 0; kt < Kt; kt += 2) {
    // ---- phase 0: compute kt from buf0; stage kt+1 (regs) -> buf1 ----
    if (active) {
      a1 = Af[(size_t)(kt + 1) * 32];
      v16h b0 = bl0[lane];
      v16h b1 = bl0[32 + lane];
      v16h b2 = bl0[64 + lane];
      v16h b3 = bl0[96 + lane];
      acc[0] = WMMA_F16(a0, b0, acc[0]);
      acc[1] = WMMA_F16(a0, b1, acc[1]);
      acc[2] = WMMA_F16(a0, b2, acc[2]);
      acc[3] = WMMA_F16(a0, b3, acc[3]);
    }
    *(uint4*)ld1 = g;
    __syncthreads();
    // ---- phase 1: compute kt+1 from buf1; prefetch kt+2; stage -> buf0 ----
    bool more = (kt + 2) < Kt;  // scalar-uniform
    if (more) {
      g = *(const uint4*)(bsrc + (size_t)(kt + 2) * 512);
      if (active) a0 = Af[(size_t)(kt + 2) * 32];
    }
    if (active) {
      v16h b0 = bl1[lane];
      v16h b1 = bl1[32 + lane];
      v16h b2 = bl1[64 + lane];
      v16h b3 = bl1[96 + lane];
      acc[0] = WMMA_F16(a1, b0, acc[0]);
      acc[1] = WMMA_F16(a1, b1, acc[1]);
      acc[2] = WMMA_F16(a1, b2, acc[2]);
      acc[3] = WMMA_F16(a1, b3, acc[3]);
    }
    if (more) *(uint4*)ld0 = g;
    __syncthreads();
  }

  if (!active) return;
  // C/D layout: row m = mt*16 + r + 8*(lane>>4), col n = nt*16 + (lane&15)
  int gg = lane >> 4, nl = lane & 15;
#pragma unroll
  for (int r = 0; r < 8; ++r) {
    int m = mt * 16 + r + 8 * gg;
    if (m >= M) continue;
    float s = cs[m], t = ct[m];
#pragma unroll
    for (int tt = 0; tt < 4; ++tt) {
      int n = (ntBase + tt) * 16 + nl;
      float v = acc[tt][r] * s + t;
      if (relu) v = fmaxf(v, 0.0f);
      out[(size_t)m * N + n] = v;
    }
  }
}

// ---------------------------------------------------------------------------
// 2x align-corners bilinear upsample + optional residual add. Input y is
// already ReLU'd (GEMM epilogue). Input H = W = 1<<lw, output 2W x 2W.
// ---------------------------------------------------------------------------
__global__ void upsample_add_kernel(const float* __restrict__ y,
                                    const float* __restrict__ add,
                                    float* __restrict__ out, int O, int lw) {
  const int W = 1 << lw;
  const int W2 = W << 1;
  size_t idx = (size_t)blockIdx.x * 256 + threadIdx.x;
  if (idx >= ((size_t)O << (2 * lw + 2))) return;
  int o = (int)(idx >> (2 * lw + 2));
  int r = (int)(idx & ((1u << (2 * lw + 2)) - 1));
  int oy = r >> (lw + 1), ox = r & (W2 - 1);
  float sc = (float)(W - 1) / (float)(W2 - 1);
  float pyv = oy * sc, pxv = ox * sc;
  int i0 = imin((int)floorf(pyv), W - 2); if (i0 < 0) i0 = 0;
  int j0 = imin((int)floorf(pxv), W - 2); if (j0 < 0) j0 = 0;
  float fy = pyv - (float)i0, fx = pxv - (float)j0;
  const float* yo = y + ((size_t)o << (2 * lw));
  float v00 = yo[(i0 << lw) + j0], v01 = yo[(i0 << lw) + j0 + 1];
  float v10 = yo[((i0 + 1) << lw) + j0], v11 = yo[((i0 + 1) << lw) + j0 + 1];
  float v = (v00 * (1.0f - fy) + v10 * fy) * (1.0f - fx) +
            (v01 * (1.0f - fy) + v11 * fy) * fx;
  out[idx] = v + (add ? add[idx] : 0.0f);
}

// ---------------------------------------------------------------------------
// Host orchestration
// ---------------------------------------------------------------------------
namespace {
struct Conv {
  _Float16* Ap;
  float* s;
  float* t;
  int M, K, Mt, Kt;
};
}  // namespace

extern "C" void kernel_launch(void* const* d_in, const int* in_sizes, int n_in,
                              void* d_out, int out_size, void* d_ws,
                              size_t ws_size, hipStream_t stream) {
  (void)in_sizes; (void)ws_size; (void)out_size;
  const int B = 4;

  const float* P[64];
  for (int i = 0; i < n_in && i < 64; ++i) P[i] = (const float*)d_in[i];
  const float* x1 = P[0];  // (4,512,32,32)
  const float* x2 = P[1];  // (4,256,64,64)
  const float* x3 = P[2];  // (4,128,128,128)
  const int hm0 = 51, wh0 = 57;  // heads: w1,b1,w2,b2,w3,b3

  size_t off = 0;
  auto alloc = [&](size_t bytes) -> void* {
    off = (off + 255) & ~(size_t)255;
    void* p = (char*)d_ws + off;
    off += bytes;
    return p;
  };

  float* x2p = (float*)alloc((size_t)B * 256 * 64 * 64 * 4);
  float* x3p = (float*)alloc((size_t)B * 128 * 128 * 128 * 4);
  float* accb = (float*)alloc((size_t)B * 256 * 64 * 64 * 4);
  float* xfeat = (float*)alloc((size_t)B * 128 * 128 * 128 * 4);
  float* hbuf = (float*)alloc((size_t)128 * 128 * 128 * 4);
  float* ybuf = (float*)alloc((size_t)128 * 64 * 64 * 4);
  float* ombuf = (float*)alloc((size_t)27 * 64 * 64 * 4);
  _Float16* colB = (_Float16*)alloc((size_t)1152 * 16384 * 2);

  auto prep_conv = [&](const float* w, const float* bias, const float* gamma,
                       const float* beta, const float* mean, const float* var,
                       int M, int K, int bn, float postmul) -> Conv {
    Conv c;
    c.M = M; c.K = K; c.Mt = (M + 15) / 16; c.Kt = K / 32;
    c.Ap = (_Float16*)alloc((size_t)c.Mt * c.Kt * 512 * 2);
    c.s = (float*)alloc((size_t)M * 4);
    c.t = (float*)alloc((size_t)M * 4);
    size_t tot = (size_t)c.Mt * c.Kt * 512;
    pack_a_kernel<<<(unsigned)CDIV(tot, 256), 256, 0, stream>>>(w, c.Ap, M, K,
                                                                c.Mt, c.Kt);
    make_coef_kernel<<<(unsigned)CDIV(M, 256), 256, 0, stream>>>(
        bias, gamma, beta, mean, var, c.s, c.t, M, bn, postmul);
    return c;
  };

  auto gemm = [&](const Conv& cv, const _Float16* Bp, float* out, int N,
                  int relu) {
    dim3 grid((unsigned)(N / 64), (unsigned)CDIV(cv.Mt, 8));
    wmma_gemm_kernel<<<grid, 256, 0, stream>>>(cv.Ap, Bp, out, cv.s, cv.t,
                                               cv.M, N, cv.Kt, cv.Mt, relu);
  };

  // --- prepack all conv weights + epilogue coefficients ---
  Conv dcnC[6], offC[6];
  for (int j = 0; j < 6; ++j) {
    const float* const* q = P + 3 + 8 * j;
    int C = (j % 2 == 0) ? 512 : 256;
    int O = (j % 2 == 0) ? 256 : 128;
    offC[j] = prep_conv(q[0], q[1], 0, 0, 0, 0, 27, C * 9, 0, 1.0f);
    dcnC[j] = prep_conv(q[2], q[3], q[4], q[5], q[6], q[7], O, C * 9, 1, 1.0f);
  }
  Conv hm1c = prep_conv(P[hm0 + 0], P[hm0 + 1], 0, 0, 0, 0, 128, 128 * 9, 0, 1.0f);
  Conv hm2c = prep_conv(P[hm0 + 2], P[hm0 + 3], 0, 0, 0, 0, 128, 128 * 9, 0, 1.0f);
  Conv hm3c = prep_conv(P[hm0 + 4], P[hm0 + 5], 0, 0, 0, 0, 80, 128, 0, 1.0f);
  Conv wh1c = prep_conv(P[wh0 + 0], P[wh0 + 1], 0, 0, 0, 0, 128, 128 * 9, 0, 1.0f);
  Conv wh2c = prep_conv(P[wh0 + 2], P[wh0 + 3], 0, 0, 0, 0, 64, 128 * 9, 0, 1.0f);
  Conv wh3c = prep_conv(P[wh0 + 4], P[wh0 + 5], 0, 0, 0, 0, 4, 64, 0, 16.0f);

  // --- DCN + BN + ReLU + up2x (+ residual), per batch ---
  auto run_dcn = [&](const float* xin, const Conv& oc, const Conv& cv,
                     const float* addsrc, float* dst, int C, int lw, int O) {
    int N = 1 << (2 * lw);
    int K = C * 9;
    size_t outN = (size_t)O * 4 * N;
    for (int b = 0; b < B; ++b) {
      const float* xb = xin + (size_t)b * C * N;
      // offset conv on the matrix pipe: im2col -> colB, om = offW @ colB
      pack_b_im2col_kernel<3><<<(unsigned)CDIV((size_t)K * N, 256), 256, 0,
                                stream>>>(xb, colB, C, lw, oc.Kt);
      gemm(oc, colB, ombuf, N, /*relu=*/0);
      deform_pack_kernel<<<(unsigned)CDIV((size_t)K * N, 256), 256, 0,
                           stream>>>(xb, ombuf, colB, C, lw, cv.Kt);
      gemm(cv, colB, ybuf, N, /*relu=*/1);
      upsample_add_kernel<<<(unsigned)CDIV(outN, 256), 256, 0, stream>>>(
          ybuf, addsrc ? addsrc + (size_t)b * outN : nullptr,
          dst + (size_t)b * outN, O, lw);
    }
  };

  auto run_conv3 = [&](const float* xin_b, const Conv& cv, float* out_b,
                       int C, int lw, int relu) {
    int N = 1 << (2 * lw);
    int K = C * 9;
    pack_b_im2col_kernel<3><<<(unsigned)CDIV((size_t)K * N, 256), 256, 0,
                              stream>>>(xin_b, colB, C, lw, cv.Kt);
    gemm(cv, colB, out_b, N, relu);
  };
  auto run_conv1 = [&](const float* xin_b, const Conv& cv, float* out_b,
                       int C, int lw, int relu) {
    int N = 1 << (2 * lw);
    pack_b_im2col_kernel<1><<<(unsigned)CDIV((size_t)C * N, 256), 256, 0,
                              stream>>>(xin_b, colB, C, lw, cv.Kt);
    gemm(cv, colB, out_b, N, relu);
  };

  // ---------------- pipeline ----------------
  run_dcn(x1, offC[0], dcnC[0], x2, x2p, 512, 5, 256);
  run_dcn(x2p, offC[1], dcnC[1], x3, x3p, 256, 6, 128);

  // hm branch
  run_dcn(x1, offC[2], dcnC[2], x2p, accb, 512, 5, 256);
  run_dcn(accb, offC[3], dcnC[3], x3p, xfeat, 256, 6, 128);
  float* hm_out = (float*)d_out;
  for (int b = 0; b < B; ++b) {
    const float* xb = xfeat + (size_t)b * 128 * 16384;
    run_conv3(xb, hm1c, hbuf, 128, 7, 1);
    run_conv3(hbuf, hm2c, hbuf, 128, 7, 1);
    run_conv1(hbuf, hm3c, hm_out + (size_t)b * 80 * 16384, 128, 7, 0);
  }

  // wh branch (reuses accb/xfeat)
  run_dcn(x1, offC[4], dcnC[4], x2p, accb, 512, 5, 256);
  run_dcn(accb, offC[5], dcnC[5], x3p, xfeat, 256, 6, 128);
  float* wh_out = (float*)d_out + (size_t)B * 80 * 16384;
  for (int b = 0; b < B; ++b) {
    const float* xb = xfeat + (size_t)b * 128 * 16384;
    run_conv3(xb, wh1c, hbuf, 128, 7, 1);
    run_conv3(hbuf, wh2c, hbuf, 128, 7, 1);
    run_conv1(hbuf, wh3c, wh_out + (size_t)b * 4 * 16384, 128, 7, 1);
  }
}
